// LSTM_43473658970462
// MI455X (gfx1250) — matrix operand order repaired and checked by reading, not compiled, not run on hardware
//
#include <hip/hip_runtime.h>
#include <hip/hip_bf16.h>

// ---------------- problem constants ----------------
#define B_    16
#define H_    512
#define E_    512
#define V_    10000
#define T_    650
#define KCAT  1536           // H + E + H   (z = [context | emb | h])
#define G4    2048           // 4*H gate width
#define NT_G  128            // G4/16  gate tiles
#define KT_G  48             // KCAT/32
#define NT_V  625            // V/16   logit tiles
#define KT_V  16             // H/32
#define NWG   64
#define NTHR  256
#define NWAVE ((NWG*NTHR)/32)   // 512 waves

// ---------------- workspace byte offsets ----------------
#define WS_BAR    0           //  2 x u32 barrier (count, generation)
#define WS_AMAX   64          // 16 x u64 packed argmax
#define WS_BG     192         // 2048 f32 combined gate bias
#define WS_Z      8448        // 16 x 1536 bf16: [0,512)=context  [1024,1536)=h
#define WS_C      57600      // 16 x 512 f32 cell state
#define WS_GATES  90368      // 16 x 2048 f32 gates
#define WS_WCAT   221440     // 128*48 tiles * 1024B swizzled bf16 [W_ih|W_hh]
#define WS_WOUT   6512896    // 625*16 tiles * 1024B swizzled bf16 W_out

typedef __attribute__((ext_vector_type(16))) __bf16   v16bf;
typedef __attribute__((ext_vector_type(8)))  float    v8f;
typedef __attribute__((ext_vector_type(4)))  unsigned v4u;
typedef __attribute__((ext_vector_type(4)))  unsigned short v4s;

union Frag { v16bf v; v4u u[2]; };

__device__ __forceinline__ unsigned short f2bf(float f) {
  unsigned u = __float_as_uint(f);
  u += 0x7FFFu + ((u >> 16) & 1u);       // round-to-nearest-even
  return (unsigned short)(u >> 16);
}
__device__ __forceinline__ float sigm(float x) { return 1.0f / (1.0f + __expf(-x)); }
__device__ __forceinline__ unsigned fmono(float f) {
  unsigned u = __float_as_uint(f);
  return (u & 0x80000000u) ? ~u : (u | 0x80000000u);   // order-preserving map
}

// ---- CDNA5 async global->LDS copy (ASYNCcnt-tracked), per ISA 08_async_tensor ----
__device__ __forceinline__ void async_ld_b128(unsigned lds_byte_addr, const void* gptr) {
  asm volatile("global_load_async_to_lds_b128 %0, %1, off"
               :: "v"(lds_byte_addr), "v"(gptr) : "memory");
}
__device__ __forceinline__ void wait_async0() {
  asm volatile("s_wait_asynccnt 0x0" ::: "memory");
}
__device__ __forceinline__ unsigned lds_addr_of(const void* p) {
  return (unsigned)(size_t)p;   // flat LDS aperture: addr[31:0] == LDS byte offset
}

// -------- grid-wide barrier (persistent kernel, all 64 WGs resident) --------
__device__ __forceinline__ void grid_barrier(unsigned* bar) {
  __threadfence();
  __syncthreads();
  if (threadIdx.x == 0) {
    unsigned gen = atomicAdd(&bar[1], 0u);
    unsigned arrived = atomicAdd(&bar[0], 1u);
    if (arrived == NWG - 1) {
      atomicExch(&bar[0], 0u);
      __threadfence();
      atomicAdd(&bar[1], 1u);
    } else {
      while (atomicAdd(&bar[1], 0u) == gen)
        __builtin_amdgcn_s_sleep(2);
    }
  }
  __syncthreads();
  __threadfence();
}

// ================= prep: swizzle [W_ih|W_hh] -> bf16 B-fragment order ========
__global__ __launch_bounds__(256) void prep_wcat(const float* __restrict__ Wih,
                                                 const float* __restrict__ Whh,
                                                 unsigned short* __restrict__ wcat) {
  int g = blockIdx.x * 256 + threadIdx.x;
  if (g >= NT_G * KT_G * 512) return;
  int tile = g >> 9, r = g & 511;
  int lane = r >> 4, q = r & 15;
  int n = tile / KT_G, k = tile % KT_G;
  int N = n * 16 + (lane & 15);
  int K = k * 32 + ((lane >> 4) << 4) + q;
  float v = (K < H_ + E_) ? Wih[(size_t)N * (H_ + E_) + K]
                          : Whh[(size_t)N * H_ + (K - (H_ + E_))];
  wcat[g] = f2bf(v);
}

__global__ __launch_bounds__(256) void prep_wout(const float* __restrict__ Wout,
                                                 unsigned short* __restrict__ wout) {
  int g = blockIdx.x * 256 + threadIdx.x;
  if (g >= NT_V * KT_V * 512) return;
  int tile = g >> 9, r = g & 511;
  int lane = r >> 4, q = r & 15;
  int n = tile / KT_V, k = tile % KT_V;
  int N = n * 16 + (lane & 15);
  int K = k * 32 + ((lane >> 4) << 4) + q;
  wout[g] = f2bf(Wout[(size_t)N * H_ + K]);
}

// ================= prep: states, biases, argmax<-sos, zero out[:,0,:] =======
__global__ __launch_bounds__(256) void prep_misc(const float* __restrict__ ctx,
                                                 const float* __restrict__ bih,
                                                 const float* __restrict__ bhh,
                                                 const int* __restrict__ sos_p,
                                                 unsigned char* __restrict__ ws,
                                                 float* __restrict__ out) {
  int g = blockIdx.x * 256 + threadIdx.x;
  unsigned short*     z    = (unsigned short*)(ws + WS_Z);
  float*              c    = (float*)(ws + WS_C);
  float*              bg   = (float*)(ws + WS_BG);
  unsigned*           bar  = (unsigned*)(ws + WS_BAR);
  unsigned long long* amax = (unsigned long long*)(ws + WS_AMAX);
  if (g < B_ * H_) {
    int b = g >> 9, j = g & (H_ - 1);
    float cv = ctx[b * H_ + j];
    z[b * KCAT + j]           = f2bf(cv);   // context slice (static)
    z[b * KCAT + H_ + E_ + j] = f2bf(cv);   // h0 = context
    c[b * H_ + j]             = cv;         // c0 = context
  }
  if (g < G4) bg[g] = bih[g] + bhh[g];
  if (g < 2)  bar[g] = 0u;
  if (g < B_) // seed argmax with <SOS> so phase-A gather picks embedding[sos]
    amax[g] = (unsigned long long)(0xFFFFFFFFu - (unsigned)sos_p[0]);
  if (g < B_ * V_) {                        // out[:, 0, :] = 0
    int b = g / V_, v = g - b * V_;
    out[(size_t)b * T_ * V_ + v] = 0.0f;
  }
}

// ======================= persistent decode kernel ===========================
__global__ __launch_bounds__(NTHR) void lstm_main(const float* __restrict__ emb,
                                                  const float* __restrict__ b_out,
                                                  unsigned char* __restrict__ ws,
                                                  float* __restrict__ out) {
  unsigned*             bar   = (unsigned*)(ws + WS_BAR);
  unsigned long long*   amax  = (unsigned long long*)(ws + WS_AMAX);
  const float*          bg    = (const float*)(ws + WS_BG);
  unsigned short*       zg    = (unsigned short*)(ws + WS_Z);
  float*                c     = (float*)(ws + WS_C);
  float*                gates = (float*)(ws + WS_GATES);
  const unsigned short* wcat  = (const unsigned short*)(ws + WS_WCAT);
  const unsigned short* wout  = (const unsigned short*)(ws + WS_WOUT);

  __shared__ unsigned short zsh[B_ * KCAT];   // 48 KB staged activation matrix

  const int tid   = threadIdx.x;
  const int gtid  = blockIdx.x * NTHR + tid;
  const int lane  = tid & 31;
  const int w     = gtid >> 5;            // global wave id, 0..511
  const int nlane = lane & 15;
  const int half  = lane >> 4;
  const bool gateWG = (blockIdx.x < NT_G / 8);   // WGs 0..15 compute gates

  // ---- one-time: stage static context slice into LDS (gate WGs only) ----
  if (gateWG) {
#pragma unroll
    for (int t = 0; t < 4; ++t) {            // 16 rows * 64 b128-chunks
      int q = tid + t * NTHR;
      int row = q >> 6, seg = q & 63;
      async_ld_b128(lds_addr_of(&zsh[row * KCAT + seg * 8]),
                    zg + row * KCAT + seg * 8);
    }
  }

#pragma unroll 1
  for (int s = 0; s < T_ - 1; ++s) {
    // ---------- Phase A: stage z (h async + emb gather), gates = z@Wcat^T ----------
    if (gateWG) {
#pragma unroll
      for (int t = 0; t < 4; ++t) {          // h slice: 1024 b128 chunks
        int q = tid + t * NTHR;
        int row = q >> 6, seg = q & 63;
        async_ld_b128(lds_addr_of(&zsh[row * KCAT + (H_ + E_) + seg * 8]),
                      zg + row * KCAT + (H_ + E_) + seg * 8);
      }
      // emb slice: gather embedding[argmax_prev] fp32 -> bf16 (amax seeded with sos)
#pragma unroll
      for (int t = 0; t < 8; ++t) {          // 2048 quads
        int q = tid + t * NTHR;
        int b = q >> 7, j4 = (q & 127) * 4;
        unsigned tok = 0xFFFFFFFFu - (unsigned)(amax[b] & 0xFFFFFFFFull);
        const float* ep = emb + (size_t)tok * E_ + j4;
        v4s pk = { f2bf(ep[0]), f2bf(ep[1]), f2bf(ep[2]), f2bf(ep[3]) };
        *(v4s*)&zsh[b * KCAT + H_ + j4] = pk;
      }
      wait_async0();
      __syncthreads();

      const int n = w;                        // one 16-wide gate tile per wave
      v8f acc = {};
      for (int k = 0; k < KT_G; ++k) {
        Frag a, b;
        const unsigned short* zr = zsh + nlane * KCAT + k * 32 + (half << 4);
        a.u[0] = *(const v4u*)(zr);           // ds_load_b128
        a.u[1] = *(const v4u*)(zr + 8);
        const unsigned short* bt = wcat + (((n * KT_G) + k) << 9) + lane * 16;
        b.u[0] = *(const v4u*)(bt);
        b.u[1] = *(const v4u*)(bt + 8);
        acc = __builtin_amdgcn_wmma_f32_16x16x32_bf16(false, a.v, false, b.v,
                                                      (short)0, acc, false, false);
      }
      const float bias = bg[n * 16 + nlane];
#pragma unroll
      for (int r = 0; r < 8; ++r) {
        int M = r + half * 8;
        gates[M * G4 + n * 16 + nlane] = acc[r] + bias;
      }
    }
    grid_barrier(bar);

    // ---------- Phase B: LSTM cell elementwise; h -> global z (bf16) ----------
    if (gtid < B_ * H_) {
      int b = gtid >> 9, j = gtid & (H_ - 1);
      const float* g = gates + b * G4;
      float ii = sigm(g[j]);
      float ff = sigm(g[H_ + j]);
      float gg = tanhf(g[2 * H_ + j]);
      float oo = sigm(g[3 * H_ + j]);
      float cn = ff * c[b * H_ + j] + ii * gg;
      float hn = oo * tanhf(cn);
      c[b * H_ + j] = cn;
      zg[b * KCAT + H_ + E_ + j] = f2bf(hn);
    }
    if (gtid < B_) amax[gtid] = 0ull;
    grid_barrier(bar);

    // ---------- Phase C: stage h to LDS, logits = h@Wout^T, fused argmax ----------
#pragma unroll
    for (int t = 0; t < 4; ++t) {            // h slice: 1024 b128 chunks
      int q = tid + t * NTHR;
      int row = q >> 6, seg = q & 63;
      async_ld_b128(lds_addr_of(&zsh[row * KCAT + (H_ + E_) + seg * 8]),
                    zg + row * KCAT + (H_ + E_) + seg * 8);
    }
    wait_async0();
    __syncthreads();

#pragma unroll 1
    for (int rep = 0; rep < 2; ++rep) {
      const int n = w + rep * NWAVE;
      if (n < NT_V) {
        v8f acc = {};
        for (int k = 0; k < KT_V; ++k) {
          Frag a, b;
          const unsigned short* zr = zsh + nlane * KCAT + (H_ + E_) + k * 32 + (half << 4);
          a.u[0] = *(const v4u*)(zr);         // ds_load_b128
          a.u[1] = *(const v4u*)(zr + 8);
          const unsigned short* bt = wout + (((n * KT_V) + k) << 9) + lane * 16;
          __builtin_prefetch(bt + 512, 0, 1); // next k tile -> global_prefetch_b8
          b.u[0] = *(const v4u*)(bt);
          b.u[1] = *(const v4u*)(bt + 8);
          acc = __builtin_amdgcn_wmma_f32_16x16x32_bf16(false, a.v, false, b.v,
                                                        (short)0, acc, false, false);
        }
        const float bias = b_out[n * 16 + nlane];
        float* orow = out + (size_t)(s + 1) * V_ + n * 16 + nlane;
#pragma unroll
        for (int r = 0; r < 8; ++r) {
          const int M = r + half * 8;
          float val = acc[r] + bias;
          orow[(size_t)M * T_ * V_] = val;
          float v = val;
          unsigned idx = (unsigned)(n * 16 + nlane);
          for (int m2 = 1; m2 < 16; m2 <<= 1) {
            float    ov = __shfl_xor(v, m2, 32);
            unsigned oi = (unsigned)__shfl_xor((int)idx, m2, 32);
            if (ov > v || (ov == v && oi < idx)) { v = ov; idx = oi; }
          }
          if (nlane == 0) {
            unsigned long long key = ((unsigned long long)fmono(v) << 32)
                                   | (unsigned long long)(0xFFFFFFFFu - idx);
            atomicMax(&amax[M], key);
          }
        }
      }
    }
    grid_barrier(bar);   // amax complete -> phase A of next step gathers embedding
  }
}

// ============================== launcher ====================================
extern "C" void kernel_launch(void* const* d_in, const int* in_sizes, int n_in,
                              void* d_out, int out_size, void* d_ws, size_t ws_size,
                              hipStream_t stream) {
  const float* ctx  = (const float*)d_in[0];
  const float* emb  = (const float*)d_in[1];
  const float* Wih  = (const float*)d_in[2];
  const float* Whh  = (const float*)d_in[3];
  const float* bih  = (const float*)d_in[4];
  const float* bhh  = (const float*)d_in[5];
  const float* Wout = (const float*)d_in[6];
  const float* bout = (const float*)d_in[7];
  const int*   sos  = (const int*)d_in[8];
  float* out = (float*)d_out;
  unsigned char* ws = (unsigned char*)d_ws;

  (void)in_sizes; (void)n_in; (void)out_size; (void)ws_size;

  prep_wcat<<<(NT_G * KT_G * 512 + 255) / 256, 256, 0, stream>>>(
      Wih, Whh, (unsigned short*)(ws + WS_WCAT));
  prep_wout<<<(NT_V * KT_V * 512 + 255) / 256, 256, 0, stream>>>(
      Wout, (unsigned short*)(ws + WS_WOUT));
  prep_misc<<<(B_ * V_ + 255) / 256, 256, 0, stream>>>(
      ctx, bih, bhh, sos, ws, out);
  lstm_main<<<NWG, NTHR, 0, stream>>>(emb, bout, ws, out);
}
